// S4Block_33706903339689
// MI455X (gfx1250) — matrix operand order repaired and checked by reading, not compiled, not run on hardware
//
#include <hip/hip_runtime.h>
#include <hip/hip_bf16.h>
#include <stddef.h>

// ---------------------------------------------------------------------------
// S4 block for (B,L,D,S) = (4,2048,1024,16) on gfx1250 (MI455X, wave32).
// All three GEMMs (in_proj, B-proj, out_proj) via v_wmma_f32_16x16x32_f16
// with f32 accumulate; 64x64 per-wave tiles for the big GEMMs (1 b128 load
// per WMMA). SSM conv is VALU with 4-way tau-split + shuffle reduction.
// ---------------------------------------------------------------------------

#define BB   4
#define LL   2048
#define DD   1024
#define SS   16
#define MM   (BB * LL)        // 8192 tokens
#define N1   (2 * DD)         // 2048 (in_proj out width)

typedef __attribute__((ext_vector_type(16))) _Float16 v16h;
typedef __attribute__((ext_vector_type(8)))  _Float16 h8;   // 16-byte load unit
typedef __attribute__((ext_vector_type(4)))  _Float16 h4;   // 8-byte store unit
typedef __attribute__((ext_vector_type(8)))  float    v8f;

// ---------------- fragment helpers (layouts per CDNA5 ISA 7.12.2) ----------

__device__ __forceinline__ v16h mk16(h8 lo, h8 hi) {
  v16h r;
#pragma unroll
  for (int i = 0; i < 8; ++i) { r[i] = lo[i]; r[i + 8] = hi[i]; }
  return r;
}

// A 16x32 f16 fragment: lane = h*16 + r holds row (m0+r),
// halves [0..7] = K{kb+8h..kb+8h+7}, [8..15] = K{kb+16+8h..}
__device__ __forceinline__ v16h load_a(const _Float16* A, int K, int m0, int kb, int r, int h) {
  const _Float16* p = A + (size_t)(m0 + r) * K + kb + 8 * h;
  return mk16(*(const h8*)p, *(const h8*)(p + 16));
}

// B 32x16 f16 fragment from row-major (N x K) weight: lane holds column (n0+r),
// halves [0..7] = K{kb+8h..kb+8h+7}, [8..15] = K{kb+16+8h..}
__device__ __forceinline__ v16h load_b(const _Float16* W, int K, int n0, int kb, int r, int h) {
  const _Float16* p = W + (size_t)(n0 + r) * K + kb + 8 * h;
  return mk16(*(const h8*)p, *(const h8*)(p + 16));
}

// 64x64 per-wave tile: 16 WMMAs per 32-deep K step, 16 b128 loads.
__device__ __forceinline__ void wmma_tile_64x64(const _Float16* A, const _Float16* W,
                                                int K, int m0, int n0, int lane,
                                                v8f acc[4][4]) {
  const int r = lane & 15, h = lane >> 4;
  for (int k = 0; k < K; k += 32) {
    v16h a[4], b[4];
#pragma unroll
    for (int i = 0; i < 4; ++i) a[i] = load_a(A, K, m0 + 16 * i, k, r, h);
#pragma unroll
    for (int j = 0; j < 4; ++j) b[j] = load_b(W, K, n0 + 16 * j, k, r, h);
#pragma unroll
    for (int i = 0; i < 4; ++i)
#pragma unroll
      for (int j = 0; j < 4; ++j)
        acc[i][j] = __builtin_amdgcn_wmma_f32_16x16x32_f16(
            false, a[i], false, b[j], (short)0, acc[i][j], false, false);
  }
}

// ---------------- small prep kernels ---------------------------------------

__global__ void __launch_bounds__(1024)
prep_kernel(const float* __restrict__ A_log, const float* __restrict__ W_B,
            const float* __restrict__ log_dt,
            float* __restrict__ dtmean, float* __restrict__ logA,
            _Float16* __restrict__ WB_h) {
  const int d = threadIdx.x;   // 0..1023
  float t = __expf(log_dt[d]);
  t = fminf(fmaxf(t, 1e-4f), 1.0f);
  __shared__ float red[1024];
  red[d] = t; __syncthreads();
  for (int s = 512; s > 0; s >>= 1) {
    if (d < s) red[d] += red[d + s];
    __syncthreads();
  }
  if (d == 0) dtmean[0] = red[0] * (1.0f / DD);
#pragma unroll
  for (int s = 0; s < SS; ++s) {
    float ab = __expf(-__expf(A_log[d * SS + s]) * t);
    ab = fminf(fmaxf(ab, 1e-8f), 1.0f - 1e-8f);
    logA[d * SS + s] = __logf(ab);
    WB_h[s * DD + d] = (_Float16)W_B[s * DD + d];
  }
}

__global__ void __launch_bounds__(256)
f32_to_f16_kernel(const float* __restrict__ src, _Float16* __restrict__ dst, int n) {
  int i = blockIdx.x * 256 + threadIdx.x;
  if (i < n) dst[i] = (_Float16)src[i];
}

// ---------------- LayerNorm -> f16 -----------------------------------------

__global__ void __launch_bounds__(256)
ln_kernel(const float* __restrict__ x, const float* __restrict__ g,
          const float* __restrict__ b, _Float16* __restrict__ xn) {
  const int m = blockIdx.x, tid = threadIdx.x;      // one token per block
  const float4 v = ((const float4*)(x + (size_t)m * DD))[tid];
  float s  = v.x + v.y + v.z + v.w;
  float sq = v.x * v.x + v.y * v.y + v.z * v.z + v.w * v.w;
  __shared__ float rs[256], rq[256];
  rs[tid] = s; rq[tid] = sq; __syncthreads();
  for (int st = 128; st > 0; st >>= 1) {
    if (tid < st) { rs[tid] += rs[tid + st]; rq[tid] += rq[tid + st]; }
    __syncthreads();
  }
  const float mu  = rs[0] * (1.0f / DD);
  const float var = rq[0] * (1.0f / DD) - mu * mu;
  const float inv = rsqrtf(var + 1e-5f);
  const int d = tid * 4;
  h4 o;
  o[0] = (_Float16)((v.x - mu) * inv * g[d + 0] + b[d + 0]);
  o[1] = (_Float16)((v.y - mu) * inv * g[d + 1] + b[d + 1]);
  o[2] = (_Float16)((v.z - mu) * inv * g[d + 2] + b[d + 2]);
  o[3] = (_Float16)((v.w - mu) * inv * g[d + 3] + b[d + 3]);
  *(h4*)(xn + (size_t)m * DD + d) = o;
}

// ---------------- GEMM1: xz = xn @ W_in^T + b_in ; split -> x_proj, silu(z) -

__global__ void __launch_bounds__(256)
gemm1_kernel(const _Float16* __restrict__ xn, const _Float16* __restrict__ Win,
             const float* __restrict__ b_in,
             float* __restrict__ xproj, _Float16* __restrict__ xproj_h,
             float* __restrict__ zbuf) {
  const int NT = N1 / 64;                      // 32 column tiles
  const int wave = blockIdx.x * 8 + (threadIdx.x >> 5);
  const int lane = threadIdx.x & 31;
  const int m0 = (wave / NT) * 64, n0 = (wave % NT) * 64;
  v8f acc[4][4] = {};
  wmma_tile_64x64(xn, Win, DD, m0, n0, lane, acc);
  const int c = lane & 15, h = lane >> 4;
#pragma unroll
  for (int i2 = 0; i2 < 4; ++i2)
#pragma unroll
    for (int j2 = 0; j2 < 4; ++j2)
#pragma unroll
      for (int i = 0; i < 8; ++i) {
        const int m = m0 + i2 * 16 + h * 8 + i;
        const int n = n0 + j2 * 16 + c;
        float v = acc[i2][j2][i] + b_in[n];
        if (n < DD) {
          xproj[(size_t)m * DD + n]   = v;
          xproj_h[(size_t)m * DD + n] = (_Float16)v;
        } else {
          zbuf[(size_t)m * DD + (n - DD)] = v / (1.0f + __expf(-v));  // silu
        }
      }
}

// ---------------- A_mean[t,s] = mean_d exp(t * logA[d,s]) ------------------

__global__ void __launch_bounds__(128)
amean_kernel(const float* __restrict__ logA, float* __restrict__ Amean) {
  const int t = blockIdx.x, tid = threadIdx.x;    // 128 threads
  const float tf = (float)t;
  float acc[SS] = {};
  for (int d = tid; d < DD; d += 128) {
    const float* la = logA + d * SS;
#pragma unroll
    for (int s = 0; s < SS; ++s) acc[s] += __expf(tf * la[s]);
  }
  __shared__ float red[128 * SS];
#pragma unroll
  for (int s = 0; s < SS; ++s) red[tid * SS + s] = acc[s];
  __syncthreads();
  for (int st = 64; st > 0; st >>= 1) {
    if (tid < st)
#pragma unroll
      for (int s = 0; s < SS; ++s) red[tid * SS + s] += red[(tid + st) * SS + s];
    __syncthreads();
  }
  if (tid < SS) Amean[t * SS + tid] = red[tid] * (1.0f / DD);
}

// ---------------- B_proj = (x_proj @ W_B^T) * mean(dt), via WMMA ------------
// M=8192, N=16 (single column tile), K=1024. W_B is (S,D) row-major == N x K.

__global__ void __launch_bounds__(256)
bproj_kernel(const _Float16* __restrict__ xproj_h, const _Float16* __restrict__ WB_h,
             const float* __restrict__ dtmean, float* __restrict__ Bproj) {
  const int wave = blockIdx.x * 8 + (threadIdx.x >> 5);
  const int lane = threadIdx.x & 31;
  const int m0 = wave * 16;                    // 512 waves cover M
  const int r = lane & 15, h = lane >> 4;
  v8f acc = {};
  for (int k = 0; k < DD; k += 32) {
    v16h a = load_a(xproj_h, DD, m0, k, r, h);
    v16h b = load_b(WB_h,    DD, 0,  k, r, h);
    acc = __builtin_amdgcn_wmma_f32_16x16x32_f16(false, a, false, b, (short)0, acc, false, false);
  }
  const float dm = dtmean[0];
#pragma unroll
  for (int i = 0; i < 8; ++i) {
    const int m = m0 + h * 8 + i;
    Bproj[(size_t)m * SS + r] = acc[i] * dm;
  }
}

// ---------------- causal conv: x_mean[b,t,s] = sum_{tau<=t} Am[t-tau]*Bp[tau]
// 4-way tau split across adjacent lanes, quad reduced with shfl_xor.

__global__ void __launch_bounds__(256)
conv_kernel(const float* __restrict__ Bproj, const float* __restrict__ Amean,
            float* __restrict__ xmean) {
  const int idx = blockIdx.x * 256 + threadIdx.x;   // 0 .. 4*MM-1
  const int part = idx & 3;
  const int bt   = idx >> 2;                        // (b,t) flattened
  const int b = bt >> 11, t = bt & (LL - 1);
  float acc[SS] = {};
  const float* bp_base = Bproj + ((size_t)b * LL) * SS;
  for (int tau = part; tau <= t; tau += 4) {
    const float4* bp = (const float4*)(bp_base + tau * SS);
    const float4* am = (const float4*)(Amean + (t - tau) * SS);
#pragma unroll
    for (int q = 0; q < 4; ++q) {
      const float4 bv = bp[q], av = am[q];
      acc[4 * q + 0] = fmaf(bv.x, av.x, acc[4 * q + 0]);
      acc[4 * q + 1] = fmaf(bv.y, av.y, acc[4 * q + 1]);
      acc[4 * q + 2] = fmaf(bv.z, av.z, acc[4 * q + 2]);
      acc[4 * q + 3] = fmaf(bv.w, av.w, acc[4 * q + 3]);
    }
  }
#pragma unroll
  for (int s = 0; s < SS; ++s) {
    acc[s] += __shfl_xor(acc[s], 1);
    acc[s] += __shfl_xor(acc[s], 2);
  }
  if (part == 0) {
    float* o = xmean + (size_t)bt * SS;
#pragma unroll
    for (int s = 0; s < SS; ++s) o[s] = acc[s];
  }
}

// ---------------- gated = (x_mean @ W_C^T + D_skip*x_proj) * z  -> f16 ------

__global__ void __launch_bounds__(256)
post_kernel(const float* __restrict__ xmean, const float* __restrict__ W_C,
            const float* __restrict__ D_skip, const float* __restrict__ xproj,
            const float* __restrict__ zbuf, _Float16* __restrict__ gated) {
  const int m = blockIdx.x, tid = threadIdx.x;     // one token per block
  __shared__ float xm[SS];
  if (tid < SS) xm[tid] = xmean[m * SS + tid];
  __syncthreads();
#pragma unroll
  for (int j = 0; j < 4; ++j) {
    const int d = tid * 4 + j;
    const float* wc = W_C + d * SS;
    float y = 0.0f;
#pragma unroll
    for (int s = 0; s < SS; ++s) y = fmaf(xm[s], wc[s], y);
    y += D_skip[d] * xproj[(size_t)m * DD + d];
    y *= zbuf[(size_t)m * DD + d];
    gated[(size_t)m * DD + d] = (_Float16)y;
  }
}

// ---------------- GEMM2: out = gated @ W_out^T + b_out + residual ----------

__global__ void __launch_bounds__(256)
gemm2_kernel(const _Float16* __restrict__ gated, const _Float16* __restrict__ Wout,
             const float* __restrict__ b_out, const float* __restrict__ x,
             float* __restrict__ out) {
  const int NT = DD / 64;                      // 16 column tiles
  const int wave = blockIdx.x * 8 + (threadIdx.x >> 5);
  const int lane = threadIdx.x & 31;
  const int m0 = (wave / NT) * 64, n0 = (wave % NT) * 64;
  v8f acc[4][4] = {};
  wmma_tile_64x64(gated, Wout, DD, m0, n0, lane, acc);
  const int c = lane & 15, h = lane >> 4;
#pragma unroll
  for (int i2 = 0; i2 < 4; ++i2)
#pragma unroll
    for (int j2 = 0; j2 < 4; ++j2)
#pragma unroll
      for (int i = 0; i < 8; ++i) {
        const int m = m0 + i2 * 16 + h * 8 + i;
        const int n = n0 + j2 * 16 + c;
        const size_t o = (size_t)m * DD + n;
        out[o] = acc[i2][j2][i] + b_out[n] + x[o];
      }
}

// ---------------- launcher --------------------------------------------------

extern "C" void kernel_launch(void* const* d_in, const int* in_sizes, int n_in,
                              void* d_out, int out_size, void* d_ws, size_t ws_size,
                              hipStream_t stream) {
  const float* x      = (const float*)d_in[0];
  const float* A_log  = (const float*)d_in[1];
  const float* W_B    = (const float*)d_in[2];
  const float* W_C    = (const float*)d_in[3];
  const float* D_skip = (const float*)d_in[4];
  const float* log_dt = (const float*)d_in[5];
  const float* W_in   = (const float*)d_in[6];
  const float* b_in   = (const float*)d_in[7];
  const float* W_out  = (const float*)d_in[8];
  const float* b_out  = (const float*)d_in[9];
  const float* ln_g   = (const float*)d_in[10];
  const float* ln_b   = (const float*)d_in[11];
  float* out = (float*)d_out;

  char* w = (char*)d_ws;
  _Float16* xn_h    = (_Float16*)w;  w += (size_t)MM * DD * 2;      // 16 MB
  _Float16* Win_h   = (_Float16*)w;  w += (size_t)N1 * DD * 2;      //  4 MB
  _Float16* Wout_h  = (_Float16*)w;  w += (size_t)DD * DD * 2;      //  2 MB
  _Float16* xproj_h = (_Float16*)w;  w += (size_t)MM * DD * 2;      // 16 MB
  float*    xproj   = (float*)w;     w += (size_t)MM * DD * 4;      // 32 MB
  float*    zbuf    = (float*)w;     w += (size_t)MM * DD * 4;      // 32 MB
  _Float16* gated   = (_Float16*)w;  w += (size_t)MM * DD * 2;      // 16 MB
  float*    Bproj   = (float*)w;     w += (size_t)MM * SS * 4;      // 512 KB
  float*    Amean   = (float*)w;     w += (size_t)LL * SS * 4;      // 128 KB
  float*    xmean   = (float*)w;     w += (size_t)MM * SS * 4;      // 512 KB
  float*    logA    = (float*)w;     w += (size_t)DD * SS * 4;      //  64 KB
  _Float16* WB_h    = (_Float16*)w;  w += (size_t)SS * DD * 2;      //  32 KB
  float*    dtmean  = (float*)w;     w += 16;

  prep_kernel<<<1, 1024, 0, stream>>>(A_log, W_B, log_dt, dtmean, logA, WB_h);
  f32_to_f16_kernel<<<(N1 * DD) / 256, 256, 0, stream>>>(W_in, Win_h, N1 * DD);
  f32_to_f16_kernel<<<(DD * DD) / 256, 256, 0, stream>>>(W_out, Wout_h, DD * DD);
  ln_kernel<<<MM, 256, 0, stream>>>(x, ln_g, ln_b, xn_h);
  gemm1_kernel<<<(MM / 64) * (N1 / 64) / 8, 256, 0, stream>>>(xn_h, Win_h, b_in,
                                                              xproj, xproj_h, zbuf);
  amean_kernel<<<LL, 128, 0, stream>>>(logA, Amean);
  bproj_kernel<<<(MM / 16) / 8, 256, 0, stream>>>(xproj_h, WB_h, dtmean, Bproj);
  conv_kernel<<<(4 * MM) / 256, 256, 0, stream>>>(Bproj, Amean, xmean);
  post_kernel<<<MM, 256, 0, stream>>>(xmean, W_C, D_skip, xproj, zbuf, gated);
  gemm2_kernel<<<(MM / 64) * (DD / 64) / 8, 256, 0, stream>>>(gated, Wout_h, b_out, x, out);
}
